// BaseAttention_29944511988362
// MI455X (gfx1250) — matrix operand order repaired and checked by reading, not compiled, not run on hardware
//
#include <hip/hip_runtime.h>
#include <hip/hip_bf16.h>
#include <math.h>

// ---------------------------------------------------------------------------
// Linear-attention transformer block for MI455X (gfx1250, wave32, WMMA).
// All heavy math via v_wmma_f32_16x16x32_bf16 (f32 accumulate).
// Fragment reads are K-contiguous -> ds_load_b128; staging kept in named
// scalars so the double-buffer pipeline lives entirely in VGPRs (no scratch).
// All fragment loads for a stage are issued before any WMMA -> one dscnt wait.
// ---------------------------------------------------------------------------

typedef __bf16 bf16;
typedef __bf16 v16bf __attribute__((ext_vector_type(16)));
typedef float  v8f   __attribute__((ext_vector_type(8)));

#define DEV __device__ __forceinline__

static constexpr int B_    = 4;
static constexpr int N_    = 4096;
static constexpr int DM    = 1024;
static constexpr int H_    = 16;
static constexpr int DEPTH = 64;
static constexpr int MROWS = B_ * N_;      // 16384 token rows
static constexpr int KTQ_SPLIT = 4;        // split-N partials for K^T Q

// ---------------- WMMA fragment helpers (contiguous k order, same in A and B) ----------------
// A operand (16x32): lane row m = lane&15, half hh = lane>>4.
//   e[i] = A[m][k0 + hh*16 + i], i = 0..15
DEV v16bf frag_a(const bf16* P, int ld, int row0, int k0, int lane) {
  const int m  = row0 + (lane & 15);
  const int hh = lane >> 4;
  union { v16bf v; bf16 e[16]; } u;
#pragma unroll
  for (int i = 0; i < 16; ++i) u.e[i] = P[m * ld + k0 + hh * 16 + i];
  return u.v;
}

// B operand (32x16): lane col n = lane&15; e[i] = B[k0 + hh*16 + i][n].
// Source tile stored TRANSPOSED [n][k] -> one contiguous 32B chunk.
DEV v16bf frag_b(const bf16* Pt, int ld, int col0, int k0, int lane) {
  const int n  = col0 + (lane & 15);
  const int hh = lane >> 4;
  union { v16bf v; bf16 e[16]; } u;
#pragma unroll
  for (int i = 0; i < 16; ++i) u.e[i] = Pt[n * ld + k0 + hh * 16 + i];
  return u.v;
}

DEV v8f wmma_bf16(v16bf a, v16bf b, v8f c) {
  return __builtin_amdgcn_wmma_f32_16x16x32_bf16(
      false, a, false, b, (short)0, c, false, false);
}

// ------------------------------- f32 -> bf16 --------------------------------
__global__ __launch_bounds__(256) void k_cvt(const float* __restrict__ s,
                                             bf16* __restrict__ d, int n4) {
  int i = blockIdx.x * blockDim.x + threadIdx.x;
  if (i >= n4) return;
  float4 f = reinterpret_cast<const float4*>(s)[i];
  union { bf16 e[4]; uint2 u; } o;
  o.e[0] = (bf16)f.x; o.e[1] = (bf16)f.y; o.e[2] = (bf16)f.z; o.e[3] = (bf16)f.w;
  reinterpret_cast<uint2*>(d)[i] = o.u;
}

// ----------------- weight transpose + convert: W[K][N] -> Wt[N][K] ----------
__global__ __launch_bounds__(256) void k_wt(const float* __restrict__ W,
                                            bf16* __restrict__ Wt) {
  __shared__ bf16 tl[64][72];
  const int bk = blockIdx.y * 64;          // source row block (k)
  const int bn = blockIdx.x * 64;          // source col block (n)
  const int t = threadIdx.x;
#pragma unroll
  for (int i = 0; i < 16; ++i) {
    const int flat = i * 256 + t;          // 64x64 tile
    const int r = flat >> 6, c = flat & 63;
    tl[r][c] = (bf16)W[(size_t)(bk + r) * DM + bn + c];
  }
  __syncthreads();
#pragma unroll
  for (int i = 0; i < 2; ++i) {
    const int flat = i * 256 + t;          // 64 rows x 8 groups
    const int r2 = flat >> 3, g = flat & 7;
    union { bf16 e[8]; uint4 u; } o;
#pragma unroll
    for (int j = 0; j < 8; ++j) o.e[j] = tl[g * 8 + j][r2];
    *reinterpret_cast<uint4*>(&Wt[(size_t)(bn + r2) * DM + bk + g * 8]) = o.u;
  }
}

// ------------------------- main tiled GEMM (bf16 WMMA) ----------------------
// C[M,N] = act(A[M,K] @ W[K,N] + bias); W supplied pre-transposed Wt[N][K].
// 128x128 block tile, BK=64, double-buffered LDS, 512 threads = 16 waves in
// an 8x2 grid; each wave owns a 16x64 strip = 4 WMMA tiles.
#define GEMM_BK 64
#define LD_S 72    // 64 + 8 halves pad (144B rows, 16B aligned)

__global__ __launch_bounds__(512, 1) void k_gemm(
    const bf16* __restrict__ A, const bf16* __restrict__ Wt,
    const float* __restrict__ bias, const float* __restrict__ resid,
    float* __restrict__ out32, bf16* __restrict__ out16,
    int M, int N, int K, int act) {
  __shared__ bf16 As [2][128 * LD_S];      // [m][k]
  __shared__ bf16 BsT[2][128 * LD_S];      // [n][k]

  const int t = threadIdx.x;
  const int lane = t & 31, wid = t >> 5;   // 16 waves
  const int waveM = wid & 7, waveN = wid >> 3;
  const int bm = blockIdx.y * 128, bn = blockIdx.x * 128;
  const int g_c8 = t & 7, g_r = t >> 3;    // 64 rows x 8 u4 groups per half

  // Named-scalar staging (promotable to VGPRs; arrays/lambdas defeat SROA).
  const bf16* pa0 = A  + (size_t)(bm + g_r) * K + g_c8 * 8;
  const bf16* pa1 = pa0 + (size_t)64 * K;
  const bf16* pb0 = Wt + (size_t)(bn + g_r) * K + g_c8 * 8;
  const bf16* pb1 = pb0 + (size_t)64 * K;
  const int soff0 = g_r * LD_S + g_c8 * 8;
  const int soff1 = soff0 + 64 * LD_S;

  v8f acc[4];
#pragma unroll
  for (int j = 0; j < 4; ++j)
#pragma unroll
    for (int r = 0; r < 8; ++r) acc[j][r] = 0.f;

  uint4 ra0 = *reinterpret_cast<const uint4*>(pa0);
  uint4 ra1 = *reinterpret_cast<const uint4*>(pa1);
  uint4 rb0 = *reinterpret_cast<const uint4*>(pb0);
  uint4 rb1 = *reinterpret_cast<const uint4*>(pb1);
  *reinterpret_cast<uint4*>(&As [0][soff0]) = ra0;
  *reinterpret_cast<uint4*>(&As [0][soff1]) = ra1;
  *reinterpret_cast<uint4*>(&BsT[0][soff0]) = rb0;
  *reinterpret_cast<uint4*>(&BsT[0][soff1]) = rb1;
  __syncthreads();

  const int nstage = K / GEMM_BK;
  for (int s = 0; s < nstage; ++s) {
    const int cur = s & 1;
    const bool more = (s + 1) < nstage;
    if (more) {
      const int kn = (s + 1) * GEMM_BK;
      ra0 = *reinterpret_cast<const uint4*>(pa0 + kn);
      ra1 = *reinterpret_cast<const uint4*>(pa1 + kn);
      rb0 = *reinterpret_cast<const uint4*>(pb0 + kn);
      rb1 = *reinterpret_cast<const uint4*>(pb1 + kn);
    }
    if (s + 2 < nstage) {   // gfx1250 global_prefetch_b8, near-cache locality
      const int kp = (s + 2) * GEMM_BK;
      __builtin_prefetch(pa0 + kp, 0, 3);
      __builtin_prefetch(pb0 + kp, 0, 3);
    }
    // Issue ALL fragment loads for both k-steps first (distinct registers),
    // then run the 8 WMMAs back-to-back: one dscnt wait per stage instead of
    // one per WMMA, and the matrix pipe overlaps the next global loads.
    {
      v16bf afs[2], bfs[2][4];
#pragma unroll
      for (int ks = 0; ks < 2; ++ks) {
        const int k0 = ks * 32;
        afs[ks] = frag_a(As[cur], LD_S, waveM * 16, k0, lane);
#pragma unroll
        for (int tn = 0; tn < 4; ++tn)
          bfs[ks][tn] = frag_b(BsT[cur], LD_S, waveN * 64 + tn * 16, k0, lane);
      }
#pragma unroll
      for (int ks = 0; ks < 2; ++ks)
#pragma unroll
        for (int tn = 0; tn < 4; ++tn)
          acc[tn] = wmma_bf16(afs[ks], bfs[ks][tn], acc[tn]);
    }
    if (more) {
      bf16* da = As [cur ^ 1];
      bf16* db = BsT[cur ^ 1];
      *reinterpret_cast<uint4*>(&da[soff0]) = ra0;
      *reinterpret_cast<uint4*>(&da[soff1]) = ra1;
      *reinterpret_cast<uint4*>(&db[soff0]) = rb0;
      *reinterpret_cast<uint4*>(&db[soff1]) = rb1;
    }
    __syncthreads();
  }

  const int hh = lane >> 4, l15 = lane & 15;
#pragma unroll
  for (int tn = 0; tn < 4; ++tn)
#pragma unroll
    for (int r = 0; r < 8; ++r) {
      const int row = bm + waveM * 16 + hh * 8 + r;
      const int col = bn + waveN * 64 + tn * 16 + l15;
      float v = acc[tn][r] + bias[col];
      if (act == 1) v = v > 0.f ? v + 1.f : __expf(v);   // elu(x)+1
      if (resid) v += resid[(size_t)row * N + col];
      if (out32) out32[(size_t)row * N + col] = v;
      else       out16[(size_t)row * N + col] = (bf16)v;
    }
}

// ------------------------- q_sum[b,h,d] = sum_n Q ---------------------------
__global__ __launch_bounds__(256) void k_qsum(const bf16* __restrict__ Qg,
                                              float* __restrict__ qsum) {
  __shared__ float red[4][64];
  const int bh = blockIdx.x, b = bh >> 4, h = bh & 15;
  const int t = threadIdx.x, d = t & 63, seg = t >> 6;
  float s = 0.f;
  for (int r = seg * (N_ / 4); r < (seg + 1) * (N_ / 4); ++r)
    s += (float)Qg[(size_t)(b * N_ + r) * DM + h * DEPTH + d];
  red[seg][d] = s;
  __syncthreads();
  if (t < 64) qsum[bh * 64 + t] = red[0][t] + red[1][t] + red[2][t] + red[3][t];
}

// ----------------- KtQ[b,h,d,e] = sum_l K[l,d] Q[l,e]  (split-N) ------------
// Both operands staged TRANSPOSED in LDS ([d][l], [e][l]) so fragment reads
// are contiguous ds_load_b128s; the l->LDS scatter is 8 b16 stores/thread.
#define KTQ_LD 40   // 32 + 8 halves pad (80B rows, 16B aligned)
__global__ __launch_bounds__(256) void k_ktq(const bf16* __restrict__ Kg,
                                             const bf16* __restrict__ Qg,
                                             float* __restrict__ ktq_part) {
  __shared__ bf16 Kts[64 * KTQ_LD];   // [d][l]
  __shared__ bf16 Qts[64 * KTQ_LD];   // [e][l]
  const int bh = blockIdx.x, b = bh >> 4, h = bh & 15;
  const int lbase = blockIdx.y * (N_ / KTQ_SPLIT);
  const int t = threadIdx.x, lane = t & 31, wid = t >> 5;
  const int row = t >> 3, c8 = t & 7;   // row = l (0..31), c8 = d-group

  v8f acc[2];
#pragma unroll
  for (int j = 0; j < 2; ++j)
#pragma unroll
    for (int r = 0; r < 8; ++r) acc[j][r] = 0.f;

  for (int lc = 0; lc < (N_ / KTQ_SPLIT) / 32; ++lc) {
    const size_t gro =
        (size_t)(b * N_ + lbase + lc * 32 + row) * DM + h * DEPTH + c8 * 8;
    union { uint4 u; bf16 e[8]; } kv, qv;
    kv.u = *reinterpret_cast<const uint4*>(&Kg[gro]);
    qv.u = *reinterpret_cast<const uint4*>(&Qg[gro]);
#pragma unroll
    for (int j = 0; j < 8; ++j) {
      Kts[(c8 * 8 + j) * KTQ_LD + row] = kv.e[j];
      Qts[(c8 * 8 + j) * KTQ_LD + row] = qv.e[j];
    }
    __syncthreads();
#pragma unroll
    for (int j = 0; j < 2; ++j) {           // 16 tiles / 8 waves = 2 each
      const int tile = wid * 2 + j;
      const int td = tile >> 2, te = tile & 3;
      v16bf a  = frag_a(Kts, KTQ_LD, td * 16, 0, lane);  // A[d][l]
      v16bf bb = frag_b(Qts, KTQ_LD, te * 16, 0, lane);  // B[l][e] via [e][l]
      acc[j] = wmma_bf16(a, bb, acc[j]);
    }
    __syncthreads();
  }
  const int hh = lane >> 4, l15 = lane & 15;
#pragma unroll
  for (int j = 0; j < 2; ++j) {
    const int tile = wid * 2 + j;
    const int td = tile >> 2, te = tile & 3;
#pragma unroll
    for (int r = 0; r < 8; ++r) {
      const int d = td * 16 + hh * 8 + r, e = te * 16 + l15;
      ktq_part[(((size_t)blockIdx.y * 64 + bh) * 64 + d) * 64 + e] = acc[j][r];
    }
  }
}

// ------- out[b,h,m,e] = (V[m,:] @ KtQ[:,e]) / (K[m,:].q_sum + eps) ----------
__global__ __launch_bounds__(256) void k_av(
    const bf16* __restrict__ Vg, const bf16* __restrict__ Kg,
    const float* __restrict__ ktq_part, const float* __restrict__ qsum,
    bf16* __restrict__ attn) {
  __shared__ bf16 BsqT[64 * LD_S];    // KtQ bf16 stored [e][d]
  __shared__ bf16 Vs[128 * LD_S];     // V rows [m][d]
  __shared__ float zs[128][2];
  __shared__ float qsl[64];
  const int bh = blockIdx.x, b = bh >> 4, h = bh & 15;
  const int m0 = blockIdx.y * 128;
  const int t = threadIdx.x, lane = t & 31, wid = t >> 5;

  if (t < 64) qsl[t] = qsum[bh * 64 + t];
#pragma unroll
  for (int i = 0; i < 16; ++i) {      // reduce split partials, stage transposed
    const int idx = i * 256 + t;
    const int d = idx >> 6, e = idx & 63;
    float v = 0.f;
#pragma unroll
    for (int sp = 0; sp < KTQ_SPLIT; ++sp)
      v += ktq_part[(((size_t)sp * 64 + bh) * 64 + d) * 64 + e];
    BsqT[e * LD_S + d] = (bf16)v;
  }
  {
    const int c8 = t & 7, r0 = t >> 3;
#pragma unroll
    for (int it = 0; it < 4; ++it) {
      const int row = r0 + it * 32;
      *reinterpret_cast<uint4*>(&Vs[row * LD_S + c8 * 8]) =
          *reinterpret_cast<const uint4*>(
              &Vg[(size_t)(b * N_ + m0 + row) * DM + h * DEPTH + c8 * 8]);
    }
  }
  __syncthreads();
  {
    const int row = t >> 1, half = t & 1;   // Z = K[m,:] . q_sum
    float s = 0.f;
    const bf16* kr = &Kg[(size_t)(b * N_ + m0 + row) * DM + h * DEPTH + half * 32];
#pragma unroll
    for (int q = 0; q < 4; ++q) {
      union { uint4 u; bf16 e[8]; } kk;
      kk.u = reinterpret_cast<const uint4*>(kr)[q];
#pragma unroll
      for (int j = 0; j < 8; ++j) s += (float)kk.e[j] * qsl[half * 32 + q * 8 + j];
    }
    zs[row][half] = s;
  }
  __syncthreads();

  v8f acc[4];
#pragma unroll
  for (int j = 0; j < 4; ++j)
#pragma unroll
    for (int r = 0; r < 8; ++r) acc[j][r] = 0.f;
#pragma unroll
  for (int ks = 0; ks < 2; ++ks) {
    v16bf a = frag_a(Vs, LD_S, wid * 16, ks * 32, lane);
#pragma unroll
    for (int tn = 0; tn < 4; ++tn) {
      v16bf bb = frag_b(BsqT, LD_S, tn * 16, ks * 32, lane);
      acc[tn] = wmma_bf16(a, bb, acc[tn]);
    }
  }
  const int hh = lane >> 4, l15 = lane & 15;
#pragma unroll
  for (int tn = 0; tn < 4; ++tn)
#pragma unroll
    for (int r = 0; r < 8; ++r) {
      const int rl = wid * 16 + hh * 8 + r;
      const float z = zs[rl][0] + zs[rl][1];
      const float v = acc[tn][r] / (z + 1e-9f);
      attn[(size_t)(b * N_ + m0 + rl) * DM + h * DEPTH + tn * 16 + l15] = (bf16)v;
    }
}

// ------------------------------- LayerNorm ----------------------------------
__global__ __launch_bounds__(256) void k_ln(const float* __restrict__ y,
                                            const float* __restrict__ gamma,
                                            const float* __restrict__ beta,
                                            float* __restrict__ out) {
  __shared__ float s1[256], s2[256];
  const size_t base = (size_t)blockIdx.x * DM;
  const int t = threadIdx.x;
  float4 v = reinterpret_cast<const float4*>(y + base)[t];
  s1[t] = v.x + v.y + v.z + v.w;
  s2[t] = v.x * v.x + v.y * v.y + v.z * v.z + v.w * v.w;
  __syncthreads();
  for (int off = 128; off > 0; off >>= 1) {
    if (t < off) { s1[t] += s1[t + off]; s2[t] += s2[t + off]; }
    __syncthreads();
  }
  const float mu = s1[0] * (1.f / DM);
  const float var = s2[0] * (1.f / DM) - mu * mu;
  const float rs = rsqrtf(var + 1e-6f);
  float4 g = reinterpret_cast<const float4*>(gamma)[t];
  float4 bb = reinterpret_cast<const float4*>(beta)[t];
  float4 o;
  o.x = (v.x - mu) * rs * g.x + bb.x;
  o.y = (v.y - mu) * rs * g.y + bb.y;
  o.z = (v.z - mu) * rs * g.z + bb.z;
  o.w = (v.w - mu) * rs * g.w + bb.w;
  reinterpret_cast<float4*>(out + base)[t] = o;
}

// ------------------------------- launcher -----------------------------------
extern "C" void kernel_launch(void* const* d_in, const int* in_sizes, int n_in,
                              void* d_out, int out_size, void* d_ws, size_t ws_size,
                              hipStream_t stream) {
  (void)in_sizes; (void)n_in; (void)out_size; (void)ws_size;
  const float* query = (const float*)d_in[0];
  const float* key   = (const float*)d_in[1];
  const float* value = (const float*)d_in[2];
  const float* Wq = (const float*)d_in[3];
  const float* bq = (const float*)d_in[4];
  const float* Wk = (const float*)d_in[5];
  const float* bk = (const float*)d_in[6];
  const float* Wv = (const float*)d_in[7];
  const float* bv = (const float*)d_in[8];
  const float* Wo = (const float*)d_in[9];
  const float* bo = (const float*)d_in[10];
  const float* gamma = (const float*)d_in[11];
  const float* beta  = (const float*)d_in[12];
  float* out = (float*)d_out;

  char* ws = (char*)d_ws;
  const size_t SZ = (size_t)MROWS * DM * 2;          // one bf16 [16384,1024]
  bf16* qb  = (bf16*)(ws + 0 * SZ);                  // bf16 staging (inputs)
  bf16* kb  = (bf16*)(ws + 1 * SZ);
  bf16* vb  = (bf16*)(ws + 2 * SZ);
  bf16* Qb  = (bf16*)(ws + 3 * SZ);                  // projected features
  bf16* Kb  = (bf16*)(ws + 4 * SZ);
  bf16* Vb  = (bf16*)(ws + 5 * SZ);
  bf16* wqb = (bf16*)(ws + 6 * SZ);                  // transposed bf16 weights
  bf16* wkb = wqb + DM * DM;
  bf16* wvb = wkb + DM * DM;
  bf16* wob = wvb + DM * DM;
  float* ktq  = (float*)(ws + 6 * SZ + (size_t)4 * DM * DM * 2);
  float* qsum = ktq + (size_t)KTQ_SPLIT * 64 * 64 * 64;
  bf16*  attn = qb;                     // reuse: qb dead after Q projection
  float* ybuf = (float*)(ws + 1 * SZ);  // reuse kb+vb: dead after K/V proj

  const int n4 = MROWS * DM / 4;
  k_cvt<<<n4 / 256, 256, 0, stream>>>(query, qb, n4);
  k_cvt<<<n4 / 256, 256, 0, stream>>>(key,   kb, n4);
  k_cvt<<<n4 / 256, 256, 0, stream>>>(value, vb, n4);
  dim3 gt(16, 16);
  k_wt<<<gt, 256, 0, stream>>>(Wq, wqb);
  k_wt<<<gt, 256, 0, stream>>>(Wk, wkb);
  k_wt<<<gt, 256, 0, stream>>>(Wv, wvb);
  k_wt<<<gt, 256, 0, stream>>>(Wo, wob);

  dim3 gg(DM / 128, MROWS / 128);   // (8, 128)
  k_gemm<<<gg, 512, 0, stream>>>(qb, wqb, bq, nullptr, nullptr, Qb, MROWS, DM, DM, 1);
  k_gemm<<<gg, 512, 0, stream>>>(kb, wkb, bk, nullptr, nullptr, Kb, MROWS, DM, DM, 1);
  k_gemm<<<gg, 512, 0, stream>>>(vb, wvb, bv, nullptr, nullptr, Vb, MROWS, DM, DM, 0);

  k_qsum<<<B_ * H_, 256, 0, stream>>>(Qb, qsum);
  k_ktq<<<dim3(B_ * H_, KTQ_SPLIT), 256, 0, stream>>>(Kb, Qb, ktq);
  k_av<<<dim3(B_ * H_, N_ / 128), 256, 0, stream>>>(Vb, Kb, ktq, qsum, attn);

  k_gemm<<<gg, 512, 0, stream>>>(attn, wob, bo, query, ybuf, nullptr, MROWS, DM, DM, 0);
  k_ln<<<MROWS, 256, 0, stream>>>(ybuf, gamma, beta, out);
}
